// MambaLiteUNet_51668456570920
// MI455X (gfx1250) — matrix operand order repaired and checked by reading, not compiled, not run on hardware
//
#include <hip/hip_runtime.h>
#include <hip/hip_bf16.h>
#include <math.h>

// ---- model dims (fixed by reference) ----
#define NBLK   4
#define DMDIM  64
#define DINNER 128
#define DST    16
#define CCH    256
#define NTOK   1024     // H*W
#define BSZ    4
#define NHEAD  8
#define HDIM   32

typedef _Float16 v16h __attribute__((ext_vector_type(16)));
typedef float    v8f  __attribute__((ext_vector_type(8)));
struct alignas(16) H8 { _Float16 v[8]; };

__device__ inline v8f zero_v8f(){ v8f z;
#pragma unroll
  for (int i=0;i<8;i++) z[i]=0.f; return z; }
__device__ inline float siluf(float x){ return x/(1.f+expf(-x)); }

// ============================================================
// Generic batched WMMA GEMM:  C = A @ op(B) (+bias) (+scale*resid)
//   A: MxK f16 row-major (lda), batch dims y,z with element strides
//   BT=1: B operand is W[N][K] row-major (out = A @ W^T)
//   BT=0: B operand is B[K][N] row-major (ldb)
//   OUTH=0: f32 output with generic (row,col) strides; OUTH=1: f16 output
//   NT:  N-tiles per wave (16xNT*16 strip) -> A fragment reused NT times.
// Out-of-range rows/cols are *clamped* for address generation (loads stay
// in-bounds, wrong data only lands in lanes whose stores are masked), so the
// K-loop has no divergent EXEC manipulation around the WMMA ops.
// ============================================================
template<int BT, int OUTH, int NT>
__global__ __launch_bounds__(256) void gemm_wmma_kernel(
    const _Float16* __restrict__ A, long long aby, long long abz, int lda,
    const _Float16* __restrict__ Bp, long long bby, long long bbz, int ldb,
    const float* __restrict__ bias,
    const float* __restrict__ resid, long long rby, long long rbz, int ldr,
    const float* __restrict__ scale_ptr, float scale_const,
    void* __restrict__ Cp, long long cby, long long cbz, long long ldcr, long long ldcc,
    int M, int N, int K)
{
  const int wave = threadIdx.x >> 5;
  const int lane = threadIdx.x & 31;
  const int strips = (N + 16*NT - 1) / (16*NT);
  const int tilesM = (M + 15) >> 4;
  int tileId = blockIdx.x * 8 + wave;
  if (tileId >= tilesM * strips) return;           // wave-uniform: EXEC stays all-ones
  const int tm = tileId / strips, tg = tileId % strips;
  const _Float16* Ab = A + (long long)blockIdx.y*aby + (long long)blockIdx.z*abz;
  const _Float16* Bb = Bp + (long long)blockIdx.y*bby + (long long)blockIdx.z*bbz;
  const int hi = lane >> 4, lr = lane & 15;
  int arow = tm*16 + lr;       if (arow >= M) arow = M - 1;   // clamp (stores masked)
  int bcol[NT];
#pragma unroll
  for (int i=0;i<NT;i++){
    int c = (tg*NT + i)*16 + lr;
    bcol[i] = (c < N) ? c : (N - 1);
  }

  v8f acc[NT];
#pragma unroll
  for (int i=0;i<NT;i++) acc[i] = zero_v8f();

  const _Float16* aptr = Ab + (long long)arow*lda + hi*16;
  for (int k0 = 0; k0 < K; k0 += 32) {
    v16h af;
    {
      if (k0 + 32 < K) __builtin_prefetch(aptr + k0 + 32, 0, 0);   // global_prefetch_b8
      H8 p0 = *(const H8*)(aptr + k0);
      H8 p1 = *(const H8*)(aptr + k0 + 8);
#pragma unroll
      for (int i=0;i<8;i++){ af[i]=p0.v[i]; af[i+8]=p1.v[i]; }
    }
    v16h bf[NT];
#pragma unroll
    for (int i=0;i<NT;i++){
      if (BT) {
        const _Float16* bp = Bb + (long long)bcol[i]*ldb + k0 + hi*16;
        H8 q0 = *(const H8*)bp;
        H8 q1 = *(const H8*)(bp + 8);
#pragma unroll
        for (int j=0;j<8;j++){ bf[i][j]=q0.v[j]; bf[i][j+8]=q1.v[j]; }
      } else {
        const _Float16* bp = Bb + (long long)(k0 + hi*16)*ldb + bcol[i];
#pragma unroll
        for (int j=0;j<16;j++) bf[i][j] = bp[(long long)j*ldb];
      }
    }
#pragma unroll
    for (int i=0;i<NT;i++)
      acc[i] = __builtin_amdgcn_wmma_f32_16x16x32_f16(false, af, false, bf[i],
                                                      (short)0, acc[i], false, false);
  }

  const float sc = scale_ptr ? scale_ptr[0] : scale_const;
#pragma unroll
  for (int i=0;i<NT;i++){
#pragma unroll
    for (int r=0;r<8;r++){
      int m = tm*16 + r + hi*8;
      int n = (tg*NT + i)*16 + lr;
      if (m < M && n < N) {
        float vv = acc[i][r];
        if (bias)  vv += bias[n];
        if (resid) vv += sc * resid[(long long)blockIdx.y*rby + (long long)blockIdx.z*rbz
                                    + (long long)m*ldr + n];
        long long off = (long long)blockIdx.y*cby + (long long)blockIdx.z*cbz
                        + (long long)m*ldcr + (long long)n*ldcc;
        if (OUTH) ((_Float16*)Cp)[off] = (_Float16)vv;
        else      ((float*)Cp)[off]    = vv;
      }
    }
  }
}

static void launch_gemm(hipStream_t st, int BT, int OUTH,
    const _Float16* A, long long aby, long long abz, int lda,
    const _Float16* B, long long bby, long long bbz, int ldb,
    const float* bias,
    const float* resid, long long rby, long long rbz, int ldr,
    const float* scale_ptr, float scale_const,
    void* C, long long cby, long long cbz, long long ldcr, long long ldcc,
    int M, int N, int K, int gy, int gz)
{
  int NT = (N % 64 == 0) ? 4 : ((N % 32 == 0) ? 2 : 1);
  int strips = (N + 16*NT - 1) / (16*NT);
  int tiles = ((M + 15)/16) * strips;
  dim3 grid((tiles + 7) / 8, gy, gz);
  dim3 block(256);
#define GDISPATCH(BTv,OHv,NTv) \
    gemm_wmma_kernel<BTv,OHv,NTv><<<grid,block,0,st>>>(A,aby,abz,lda,B,bby,bbz,ldb,\
        bias,resid,rby,rbz,ldr,scale_ptr,scale_const,C,cby,cbz,ldcr,ldcc,M,N,K)
  if (BT) {
    if (OUTH) { if (NT==4) GDISPATCH(1,1,4); else if (NT==2) GDISPATCH(1,1,2); else GDISPATCH(1,1,1); }
    else      { if (NT==4) GDISPATCH(1,0,4); else if (NT==2) GDISPATCH(1,0,2); else GDISPATCH(1,0,1); }
  } else {
    if (OUTH) { if (NT==4) GDISPATCH(0,1,4); else if (NT==2) GDISPATCH(0,1,2); else GDISPATCH(0,1,1); }
    else      { if (NT==4) GDISPATCH(0,0,4); else if (NT==2) GDISPATCH(0,0,2); else GDISPATCH(0,0,1); }
  }
#undef GDISPATCH
}

// ============================================================ helpers
__global__ void f32_to_f16_kernel(const float* __restrict__ in, _Float16* __restrict__ out, long long n)
{
  long long i = (long long)blockIdx.x*blockDim.x + threadIdx.x;
  if (i < n) out[i] = (_Float16)in[i];
}

// Row layernorm over C=256 (one block per row); optional raw transpose copy,
// optional exact-GELU, optional f32 and/or f16 outputs (B,N,C) contiguous.
__global__ __launch_bounds__(256) void row_ln_kernel(
    const float* __restrict__ in, long long sb, long long sn, long long scs,
    const float* __restrict__ gamma, const float* __restrict__ beta,
    float* __restrict__ raw_out, float* __restrict__ f32_out, _Float16* __restrict__ h_out,
    int gelu_flag)
{
  int row = blockIdx.x;
  int b = row / NTOK, n = row % NTOK;
  int t = threadIdx.x;
  __shared__ float red[256];
  float val = in[(long long)b*sb + (long long)n*sn + (long long)t*scs];
  if (raw_out) raw_out[(long long)row*CCH + t] = val;
  red[t] = val; __syncthreads();
  for (int s=128;s>0;s>>=1){ if (t<s) red[t]+=red[t+s]; __syncthreads(); }
  float mu = red[0] / (float)CCH; __syncthreads();
  float d = val - mu;
  red[t] = d*d; __syncthreads();
  for (int s=128;s>0;s>>=1){ if (t<s) red[t]+=red[t+s]; __syncthreads(); }
  float rs = rsqrtf(red[0]/(float)CCH + 1e-5f);
  float y = d * rs;
  if (gamma) y = y*gamma[t] + beta[t];
  if (gelu_flag) y = 0.5f*y*(1.f + erff(y*0.70710678118654752f));
  if (f32_out) f32_out[(long long)row*CCH + t] = y;
  if (h_out)   h_out[(long long)row*CCH + t]   = (_Float16)y;
}

// Mamba depthwise causal conv (K=4, left pad 3) + SiLU; u from uz[...,:128]
__global__ void mamba_conv_kernel(const float* __restrict__ uz,
    const float* __restrict__ convw, const float* __restrict__ convb,
    float* __restrict__ u32, _Float16* __restrict__ u16)
{
  int idx = blockIdx.x*blockDim.x + threadIdx.x;
  if (idx >= NBLK*BSZ*NTOK*DINNER) return;
  int d  = idx & 127;
  int n  = (idx >> 7) & 1023;
  int bb = idx >> 17;            // nb*BSZ + b
  int nb = bb >> 2;
  float acc = convb[nb*DINNER + d];
#pragma unroll
  for (int k=0;k<4;k++){
    int nn = n - 3 + k;
    if (nn >= 0) acc += convw[(nb*DINNER + d)*4 + k] * uz[((long long)bb*NTOK + nn)*256 + d];
  }
  float s = siluf(acc);
  u32[idx] = s; u16[idx] = (_Float16)s;
}

// Selective scan: one block per (nb,b); one lane per d-channel; 16 states in regs.
__global__ __launch_bounds__(128) void scan_kernel(
    const float* __restrict__ xdbc, const float* __restrict__ uz, const float* __restrict__ u32,
    const float* __restrict__ dtw, const float* __restrict__ dtb,
    const float* __restrict__ alog, const float* __restrict__ Dp,
    _Float16* __restrict__ ym)
{
  int bb = blockIdx.x;           // nb*BSZ + b
  int nb = bb >> 2;
  int d  = threadIdx.x;
  float A_s[16], h[16];
#pragma unroll
  for (int s=0;s<16;s++){ A_s[s] = -expf(alog[(nb*DINNER + d)*16 + s]); h[s] = 0.f; }
  float w0 = dtw[(nb*DINNER+d)*4+0], w1 = dtw[(nb*DINNER+d)*4+1];
  float w2 = dtw[(nb*DINNER+d)*4+2], w3 = dtw[(nb*DINNER+d)*4+3];
  float db = dtb[nb*DINNER + d];
  float Dv = Dp [nb*DINNER + d];
  __shared__ float sx[36];
  for (int n=0;n<NTOK;n++){
    __syncthreads();
    if (d < 36) sx[d] = xdbc[((long long)bb*NTOK + n)*36 + d];
    __syncthreads();
    float dtv = sx[0]*w0 + sx[1]*w1 + sx[2]*w2 + sx[3]*w3 + db;
    dtv = (dtv > 20.f) ? dtv : log1pf(expf(dtv));         // softplus
    float uval = u32[((long long)bb*NTOK + n)*DINNER + d];
    float dtu = dtv * uval;
    float y = 0.f;
#pragma unroll
    for (int s=0;s<16;s++){
      h[s] = expf(dtv * A_s[s]) * h[s] + dtu * sx[4+s];   // B
      y   += h[s] * sx[20+s];                             // C
    }
    float z = uz[((long long)bb*NTOK + n)*256 + DINNER + d];
    ym[((long long)bb*NTOK + n)*DINNER + d] = (_Float16)((y + uval*Dv) * siluf(z));
  }
}

// Gate path: grouped depthwise (4 groups of 64ch, k=3, pad 1) -> (B,4,N)
__global__ void gate_dw_kernel(const float* __restrict__ xcat,
    const float* __restrict__ qdw_w, const float* __restrict__ qdw_b, float* __restrict__ dwq)
{
  int idx = blockIdx.x*blockDim.x + threadIdx.x;
  if (idx >= BSZ*4*NTOK) return;
  int n = idx & 1023, g = (idx>>10)&3, b = idx>>12;
  float acc = qdw_b[g];
  for (int k=0;k<3;k++){
    int nn = n + k - 1; if (nn < 0 || nn >= NTOK) continue;
    const float* xr = xcat + ((long long)(b*NTOK + nn))*CCH + g*64;
#pragma unroll
    for (int i=0;i<64;i++) acc += qdw_w[(g*64+i)*3 + k] * xr[i];
  }
  dwq[idx] = acc;   // (b,g,n)
}

// pointwise 4->4 + sigmoid gate applied per 64-channel block
__global__ void gate_apply_kernel(const float* __restrict__ xcat, const float* __restrict__ dwq,
    const float* __restrict__ qpw_w, const float* __restrict__ qpw_b, float* __restrict__ xg)
{
  int idx = blockIdx.x*blockDim.x + threadIdx.x;
  if (idx >= BSZ*NTOK*CCH) return;
  int c = idx & 255, n = (idx>>8)&1023, b = idx>>18;
  int nb = c >> 6;
  float q = qpw_b[nb];
#pragma unroll
  for (int g=0;g<4;g++) q += qpw_w[nb*4 + g] * dwq[(b*4 + g)*NTOK + n];
  xg[idx] = (1.f/(1.f + expf(-q))) * xcat[idx];
}

// depthwise conv over sequence (k=3,pad 1) on (B,N,Ctot); channel j<C0 from in0 else in1
__global__ void dwn_kernel(const float* __restrict__ in0, const float* __restrict__ in1,
    int C0, int Ctot, const float* __restrict__ w, const float* __restrict__ bias,
    _Float16* __restrict__ outh, int out_ld, int out_off)
{
  long long idx = (long long)blockIdx.x*blockDim.x + threadIdx.x;
  if (idx >= (long long)BSZ*NTOK*Ctot) return;
  int j = (int)(idx % Ctot);
  int n = (int)((idx / Ctot) % NTOK);
  int b = (int)(idx / ((long long)Ctot*NTOK));
  float acc = bias[j];
#pragma unroll
  for (int k=0;k<3;k++){
    int nn = n + k - 1; if (nn < 0 || nn >= NTOK) continue;
    float v = (j < C0) ? in0[((long long)(b*NTOK + nn))*C0 + j]
                       : in1[((long long)(b*NTOK + nn))*(Ctot - C0) + (j - C0)];
    acc += w[j*3 + k] * v;
  }
  outh[((long long)(b*NTOK + n))*out_ld + out_off + j] = (_Float16)acc;
}

// qkv (B,N,768) f16 -> q/k/v (B,NH,N,32) f16; q pre-scaled by 1/sqrt(hd)
__global__ void qkv_repack_kernel(const _Float16* __restrict__ qkv,
    _Float16* __restrict__ q, _Float16* __restrict__ k, _Float16* __restrict__ v)
{
  long long idx = (long long)blockIdx.x*blockDim.x + threadIdx.x;
  if (idx >= (long long)BSZ*NTOK*768) return;
  int col = (int)(idx % 768);
  long long bn = idx / 768;
  int n = (int)(bn % NTOK), b = (int)(bn / NTOK);
  int t = col >> 8, h = (col & 255) >> 5, j = col & 31;
  _Float16 val = qkv[idx];
  long long dst = (((long long)b*NHEAD + h)*NTOK + n)*HDIM + j;
  if (t == 0)      q[dst] = (_Float16)((float)val * 0.17677669529663687f);
  else if (t == 1) k[dst] = val;
  else             v[dst] = val;
}

// softmax over 1024 columns (one block per row), write f16 probs
__global__ __launch_bounds__(256) void softmax_kernel(const float* __restrict__ S, _Float16* __restrict__ P)
{
  int row = blockIdx.x, t = threadIdx.x;
  __shared__ float red[256];
  const float* sr = S + (long long)row*NTOK;
  float v[4]; float mx = -1e30f;
#pragma unroll
  for (int i=0;i<4;i++){ v[i] = sr[t + i*256]; mx = fmaxf(mx, v[i]); }
  red[t] = mx; __syncthreads();
  for (int s=128;s>0;s>>=1){ if (t<s) red[t]=fmaxf(red[t],red[t+s]); __syncthreads(); }
  mx = red[0]; __syncthreads();
  float sum = 0.f;
#pragma unroll
  for (int i=0;i<4;i++){ v[i] = expf(v[i]-mx); sum += v[i]; }
  red[t] = sum; __syncthreads();
  for (int s=128;s>0;s>>=1){ if (t<s) red[t]+=red[t+s]; __syncthreads(); }
  float inv = 1.f/red[0];
  _Float16* pr = P + (long long)row*NTOK;
#pragma unroll
  for (int i=0;i<4;i++) pr[t + i*256] = (_Float16)(v[i]*inv);
}

// instance-norm stats over H*W per (b,c): store mu, rsqrt(var+eps)
__global__ __launch_bounds__(256) void inorm_stats_kernel(const float* __restrict__ t1, float* __restrict__ stats)
{
  int c = blockIdx.x & 255, b = blockIdx.x >> 8, t = threadIdx.x;
  __shared__ float r1[256], r2[256];
  float s = 0.f, s2 = 0.f;
#pragma unroll
  for (int i=0;i<4;i++){
    int n = t + i*256;
    float v = t1[((long long)(b*NTOK + n))*CCH + c];
    s += v; s2 += v*v;
  }
  r1[t] = s; r2[t] = s2; __syncthreads();
  for (int k=128;k>0;k>>=1){ if (t<k){ r1[t]+=r1[t+k]; r2[t]+=r2[t+k]; } __syncthreads(); }
  if (t == 0){
    float mu  = r1[0] / (float)NTOK;
    float var = r2[0] / (float)NTOK - mu*mu;
    stats[blockIdx.x*2]   = mu;
    stats[blockIdx.x*2+1] = rsqrtf(var + 1e-5f);
  }
}

// 3x3 depthwise conv on instance-normalized (B,C,H,W)
__global__ void dwc_kernel(const float* __restrict__ t1, const float* __restrict__ stats,
    const float* __restrict__ w, const float* __restrict__ bias, float* __restrict__ out2)
{
  int idx = blockIdx.x*blockDim.x + threadIdx.x;
  if (idx >= BSZ*CCH*NTOK) return;
  int wc = idx & 31, hr = (idx>>5)&31, c = (idx>>10)&255, b = idx>>18;
  float mu = stats[(b*256 + c)*2], rs = stats[(b*256 + c)*2 + 1];
  float acc = bias[c];
#pragma unroll
  for (int dy=0;dy<3;dy++){
    int hy = hr + dy - 1; if (hy < 0 || hy >= 32) continue;
#pragma unroll
    for (int dx=0;dx<3;dx++){
      int wx = wc + dx - 1; if (wx < 0 || wx >= 32) continue;
      int n = hy*32 + wx;
      acc += w[c*9 + dy*3 + dx] * ((t1[((long long)(b*NTOK + n))*CCH + c] - mu) * rs);
    }
  }
  out2[((long long)b*CCH + c)*NTOK + (hr*32 + wc)] = acc;
}

// ============================================================
extern "C" void kernel_launch(void* const* d_in, const int* in_sizes, int n_in,
                              void* d_out, int out_size, void* d_ws, size_t ws_size,
                              hipStream_t stream)
{
  (void)in_sizes; (void)n_in; (void)out_size; (void)ws_size;
  const float* x        = (const float*)d_in[0];
  const float* norm_w   = (const float*)d_in[1];
  const float* norm_b   = (const float*)d_in[2];
  const float* skip_s   = (const float*)d_in[3];
  const float* m_inw    = (const float*)d_in[4];
  const float* m_convw  = (const float*)d_in[5];
  const float* m_convb  = (const float*)d_in[6];
  const float* m_xpw    = (const float*)d_in[7];
  const float* m_dtw    = (const float*)d_in[8];
  const float* m_dtb    = (const float*)d_in[9];
  const float* m_alog   = (const float*)d_in[10];
  const float* m_d      = (const float*)d_in[11];
  const float* m_outw   = (const float*)d_in[12];
  const float* qdw_w    = (const float*)d_in[13];
  const float* qdw_b    = (const float*)d_in[14];
  const float* qpw_w    = (const float*)d_in[15];
  const float* qpw_b    = (const float*)d_in[16];
  const float* rdw_w    = (const float*)d_in[17];
  const float* rdw_b    = (const float*)d_in[18];
  const float* rpw_w    = (const float*)d_in[19];
  const float* rpw_b    = (const float*)d_in[20];
  const float* fdw_w    = (const float*)d_in[21];
  const float* fdw_b    = (const float*)d_in[22];
  const float* fpw_w    = (const float*)d_in[23];
  const float* fpw_b    = (const float*)d_in[24];
  const float* lc_w     = (const float*)d_in[25];
  const float* lc_b     = (const float*)d_in[26];
  const float* attn_inw = (const float*)d_in[27];
  const float* attn_inb = (const float*)d_in[28];
  const float* attn_outw= (const float*)d_in[29];
  const float* attn_outb= (const float*)d_in[30];
  const float* lgn_w    = (const float*)d_in[31];
  const float* lgn_b    = (const float*)d_in[32];
  const float* fc_w     = (const float*)d_in[33];
  const float* fc_b     = (const float*)d_in[34];
  const float* dwc_w    = (const float*)d_in[35];
  const float* dwc_b    = (const float*)d_in[36];
  const float* proj_w   = (const float*)d_in[37];
  const float* proj_b   = (const float*)d_in[38];

  size_t off = 0;
  auto carve = [&](size_t bytes) -> void* {
    off = (off + 255) & ~(size_t)255;
    void* p = (char*)d_ws + off;
    off += bytes;
    return p;
  };
  const long long BNC = (long long)BSZ*NTOK*CCH;          // 1,048,576

  _Float16* w_inw16  = (_Float16*)carve(sizeof(_Float16)*NBLK*256*DMDIM);
  _Float16* w_xpw16  = (_Float16*)carve(sizeof(_Float16)*NBLK*36*DINNER);
  _Float16* w_outw16 = (_Float16*)carve(sizeof(_Float16)*NBLK*DMDIM*DINNER);
  _Float16* w_rpw16  = (_Float16*)carve(sizeof(_Float16)*256*256);
  _Float16* w_fpw16  = (_Float16*)carve(sizeof(_Float16)*256*512);
  _Float16* w_ai16   = (_Float16*)carve(sizeof(_Float16)*768*256);
  _Float16* w_ao16   = (_Float16*)carve(sizeof(_Float16)*256*256);
  _Float16* w_fc16   = (_Float16*)carve(sizeof(_Float16)*256*512);
  _Float16* w_pr16   = (_Float16*)carve(sizeof(_Float16)*256*256);
  float*    xs    = (float*)carve(sizeof(float)*BNC);
  float*    xn32  = (float*)carve(sizeof(float)*BNC);
  _Float16* xn16  = (_Float16*)carve(sizeof(_Float16)*BNC);
  float*    uz    = (float*)carve(sizeof(float)*(long long)16*NTOK*256);
  float*    u32   = (float*)carve(sizeof(float)*(long long)16*NTOK*128);
  _Float16* u16   = (_Float16*)carve(sizeof(_Float16)*(long long)16*NTOK*128);
  float*    xdbc  = (float*)carve(sizeof(float)*(long long)16*NTOK*36);
  _Float16* ym16  = (_Float16*)carve(sizeof(_Float16)*(long long)16*NTOK*128);
  float*    xcat  = (float*)carve(sizeof(float)*BNC);
  float*    dwq   = (float*)carve(sizeof(float)*BSZ*4*NTOK);
  float*    xg    = (float*)carve(sizeof(float)*BNC);
  _Float16* rd16  = (_Float16*)carve(sizeof(_Float16)*BNC);
  float*    Rb    = (float*)carve(sizeof(float)*BNC);
  _Float16* fd16  = (_Float16*)carve(sizeof(_Float16)*BNC*2);
  float*    xm2   = (float*)carve(sizeof(float)*BNC);
  _Float16* xm2h  = (_Float16*)carve(sizeof(_Float16)*BNC);
  _Float16* qkvh  = (_Float16*)carve(sizeof(_Float16)*(long long)BSZ*NTOK*768);
  _Float16* qh    = (_Float16*)carve(sizeof(_Float16)*BNC);
  _Float16* kh    = (_Float16*)carve(sizeof(_Float16)*BNC);
  _Float16* vh    = (_Float16*)carve(sizeof(_Float16)*BNC);
  float*    scor  = (float*)carve(sizeof(float)*(long long)NTOK*NTOK);
  _Float16* atth  = (_Float16*)carve(sizeof(_Float16)*(long long)NTOK*NTOK);
  _Float16* oh    = (_Float16*)carve(sizeof(_Float16)*BNC);
  _Float16* hcat  = (_Float16*)carve(sizeof(_Float16)*BNC*2);
  float*    xo    = (float*)carve(sizeof(float)*BNC);
  float*    t1    = (float*)carve(sizeof(float)*BNC);
  float*    stats = (float*)carve(sizeof(float)*BSZ*CCH*2);
  float*    out2  = (float*)carve(sizeof(float)*BNC);
  _Float16* xmixh = (_Float16*)carve(sizeof(_Float16)*BNC);

  auto cvt = [&](const float* s, _Float16* d, long long n){
    f32_to_f16_kernel<<<(unsigned)((n+255)/256),256,0,stream>>>(s,d,n);
  };
  // -- weight conversions (f32 -> f16) --
  cvt(m_inw,    w_inw16,  (long long)NBLK*256*DMDIM);
  cvt(m_xpw,    w_xpw16,  (long long)NBLK*36*DINNER);
  cvt(m_outw,   w_outw16, (long long)NBLK*DMDIM*DINNER);
  cvt(rpw_w,    w_rpw16,  256*256);
  cvt(fpw_w,    w_fpw16,  256*512);
  cvt(attn_inw, w_ai16,   768*256);
  cvt(attn_outw,w_ao16,   256*256);
  cvt(fc_w,     w_fc16,   256*512);
  cvt(proj_w,   w_pr16,   256*256);

  // -- xs = transpose(x); xn = LN(xs, norm) --
  row_ln_kernel<<<BSZ*NTOK,256,0,stream>>>(x, (long long)CCH*NTOK, 1, NTOK,
      norm_w, norm_b, xs, xn32, xn16, 0);

  // -- mamba in-proj: uz[nb,b,n,0:256] = xn_block @ inw^T --
  launch_gemm(stream,1,0, xn16, 64, (long long)NTOK*CCH, CCH,
      w_inw16, (long long)256*DMDIM, 0, DMDIM,
      nullptr, nullptr,0,0,0, nullptr,0.f,
      uz, (long long)BSZ*NTOK*256, (long long)NTOK*256, 256, 1,
      NTOK, 256, DMDIM, NBLK, BSZ);

  mamba_conv_kernel<<<(NBLK*BSZ*NTOK*DINNER)/256,256,0,stream>>>(uz, m_convw, m_convb, u32, u16);

  // -- x-proj: xdbc = u @ xpw^T (N=36, NT=1 path, guarded stores) --
  launch_gemm(stream,1,0, u16, (long long)BSZ*NTOK*DINNER, (long long)NTOK*DINNER, DINNER,
      w_xpw16, (long long)36*DINNER, 0, DINNER,
      nullptr, nullptr,0,0,0, nullptr,0.f,
      xdbc, (long long)BSZ*NTOK*36, (long long)NTOK*36, 36, 1,
      NTOK, 36, DINNER, NBLK, BSZ);

  scan_kernel<<<16,128,0,stream>>>(xdbc, uz, u32, m_dtw, m_dtb, m_alog, m_d, ym16);

  // -- out-proj + skip: xcat[b,n,nb*64+.] = ym @ outw^T + skip_scale*xn --
  launch_gemm(stream,1,0, ym16, (long long)BSZ*NTOK*DINNER, (long long)NTOK*DINNER, DINNER,
      w_outw16, (long long)DMDIM*DINNER, 0, DINNER,
      nullptr, xn32, 64, (long long)NTOK*CCH, CCH, skip_s, 0.f,
      xcat, 64, (long long)NTOK*CCH, CCH, 1,
      NTOK, DMDIM, DINNER, NBLK, BSZ);

  // -- gating --
  gate_dw_kernel<<<(BSZ*4*NTOK)/256,256,0,stream>>>(xcat, qdw_w, qdw_b, dwq);
  gate_apply_kernel<<<(unsigned)(BNC/256),256,0,stream>>>(xcat, dwq, qpw_w, qpw_b, xg);

  // -- R = rpw(rdw(xg)) + xg --
  dwn_kernel<<<(unsigned)(BNC/256),256,0,stream>>>(xg, xg, 256, 256, rdw_w, rdw_b, rd16, 256, 0);
  launch_gemm(stream,1,0, rd16, 0,0,256, w_rpw16, 0,0,256, rpw_b,
      xg, 0,0,256, nullptr,1.0f, Rb, 0,0,256,1, BSZ*NTOK, 256, 256, 1,1);

  // -- xm2 = fpw(fdw([xs,R])) + xs --
  dwn_kernel<<<(unsigned)(2*BNC/256),256,0,stream>>>(xs, Rb, 256, 512, fdw_w, fdw_b, fd16, 512, 0);
  launch_gemm(stream,1,0, fd16, 0,0,512, w_fpw16, 0,0,512, fpw_b,
      xs, 0,0,256, nullptr,1.0f, xm2, 0,0,256,1, BSZ*NTOK, 256, 512, 1,1);
  cvt(xm2, xm2h, BNC);

  // -- MHA --
  launch_gemm(stream,1,1, xm2h, 0,0,256, w_ai16, 0,0,256, attn_inb,
      nullptr,0,0,0, nullptr,0.f, qkvh, 0,0,768,1, BSZ*NTOK, 768, 256, 1,1);
  qkv_repack_kernel<<<(unsigned)(((long long)BSZ*NTOK*768)/256),256,0,stream>>>(qkvh, qh, kh, vh);
  for (int b=0;b<BSZ;b++)
    for (int h=0;h<NHEAD;h++){
      long long o = (((long long)b*NHEAD + h)*NTOK)*HDIM;
      launch_gemm(stream,1,0, qh+o, 0,0,HDIM, kh+o, 0,0,HDIM,
          nullptr, nullptr,0,0,0, nullptr,0.f, scor, 0,0, NTOK,1, NTOK, NTOK, HDIM, 1,1);
      softmax_kernel<<<NTOK,256,0,stream>>>(scor, atth);
      launch_gemm(stream,0,1, atth, 0,0,NTOK, vh+o, 0,0,HDIM,
          nullptr, nullptr,0,0,0, nullptr,0.f,
          oh + (long long)b*NTOK*CCH + h*HDIM, 0,0, CCH,1, NTOK, HDIM, NTOK, 1,1);
    }
  // attention out-proj -> hcat[:,256:512] (f16)
  launch_gemm(stream,1,1, oh, 0,0,256, w_ao16, 0,0,256, attn_outb,
      nullptr,0,0,0, nullptr,0.f, hcat + 256, 0,0, 512,1, BSZ*NTOK, 256, 256, 1,1);
  // x_local -> hcat[:,0:256] (f16)
  dwn_kernel<<<(unsigned)(BNC/256),256,0,stream>>>(xm2, xm2, 256, 256, lc_w, lc_b, hcat, 512, 0);

  // -- fc over concat --
  launch_gemm(stream,1,0, hcat, 0,0,512, w_fc16, 0,0,512, fc_b,
      nullptr,0,0,0, nullptr,0.f, xo, 0,0,256,1, BSZ*NTOK, 256, 512, 1,1);

  // -- LN(lgn)+GELU; instance norm stats; 3x3 depthwise --
  row_ln_kernel<<<BSZ*NTOK,256,0,stream>>>(xo, (long long)NTOK*CCH, CCH, 1,
      lgn_w, lgn_b, nullptr, t1, nullptr, 1);
  inorm_stats_kernel<<<BSZ*CCH,256,0,stream>>>(t1, stats);
  dwc_kernel<<<(unsigned)(BNC/256),256,0,stream>>>(t1, stats, dwc_w, dwc_b, out2);

  // -- final LN(norm) + projection, written transposed into d_out (B,COUT,H,W) --
  row_ln_kernel<<<BSZ*NTOK,256,0,stream>>>(out2, (long long)CCH*NTOK, 1, NTOK,
      norm_w, norm_b, nullptr, nullptr, xmixh, 0);
  launch_gemm(stream,1,0, xmixh, 0, (long long)NTOK*CCH, 256,
      w_pr16, 0,0,256, proj_b, nullptr,0,0,0, nullptr,0.f,
      d_out, 0, (long long)CCH*NTOK, 1, NTOK,
      NTOK, CCH, CCH, 1, BSZ);
}
